// MultiHeadAttention_20134806683812
// MI455X (gfx1250) — compile-verified, matrix operand order checked by reading, not run on hardware
//
#include <hip/hip_runtime.h>
#include <hip/hip_bf16.h>

// ---------------------------------------------------------------------------
// Multi-head attention forward for MI455X (gfx1250), bf16 WMMA pipeline:
//   K0: one-shot fp32 -> bf16 conversion of X and all weight matrices
//   K1: QKV = X @ W{q,k,v}^T   (bf16 WMMA, async-LDS double-buffered weights)
//   K2: flash attention        (online softmax, K/V tiles async-staged in LDS)
//   K3: out = ctx @ Wo^T + bo  (bf16 WMMA, fp32 out)
// ---------------------------------------------------------------------------

#define EMBED   768
#define SEQ     4096
#define BATCH   2
#define NHEADS  12
#define HDIM    64
#define ROWS    (BATCH * SEQ)   // 8192
#define NKT     (EMBED / 32)    // 24 k-steps
#define NJT     (SEQ / 64)      // 64 j-tiles

typedef __bf16 bf16_t;
typedef __attribute__((ext_vector_type(16))) __bf16       v16bf;
typedef __attribute__((ext_vector_type(8)))  float        v8f;
typedef __attribute__((ext_vector_type(4)))  float        v4f;
typedef __attribute__((ext_vector_type(4)))  unsigned int v4u;
typedef __attribute__((ext_vector_type(4)))  int          v4i;

union Frag16 { v16bf v; v4u u[2]; };
union H8     { bf16_t h[8]; v4u u; };

__device__ __forceinline__ v8f wmma_bf16(v16bf a, v16bf b, v8f c) {
  return __builtin_amdgcn_wmma_f32_16x16x32_bf16(false, a, false, b,
                                                 (short)0, c, false, false);
}

__device__ __forceinline__ v16bf frag_ld_bf16(const bf16_t* p0, const bf16_t* p1) {
  Frag16 f;
  f.u[0] = *(const v4u*)p0;
  f.u[1] = *(const v4u*)p1;
  return f.v;
}

// ---- CDNA5 async global->LDS copy (ASYNCcnt), with portable fallback ------
#if __has_builtin(__builtin_amdgcn_global_load_async_to_lds_b128)
#define HAS_ASYNC_LDS 1
#else
#define HAS_ASYNC_LDS 0
#endif

__device__ __forceinline__ void async_cp16(bf16_t* dst_lds, const bf16_t* src) {
#if HAS_ASYNC_LDS
  // signature (from hipcc diagnostic): (int4 AS1*, int4 AS3*, imm offset, imm cpol)
  __builtin_amdgcn_global_load_async_to_lds_b128(
      (__attribute__((address_space(1))) v4i*)src,
      (__attribute__((address_space(3))) v4i*)dst_lds, 0, 0);
#else
  *(v4u*)dst_lds = *(const v4u*)src;
#endif
}

__device__ __forceinline__ void async_wait0() {
#if HAS_ASYNC_LDS
#if __has_builtin(__builtin_amdgcn_s_wait_asynccnt)
  __builtin_amdgcn_s_wait_asynccnt(0);
#else
  asm volatile("s_wait_asynccnt 0x0" ::: "memory");
#endif
#endif
}

// ---------------------------------------------------------------------------
// Kernel 0: elementwise fp32 -> bf16 (8 elements / thread, b128 in, b128 out)
// ---------------------------------------------------------------------------
__global__ __launch_bounds__(256) void cvt_bf16_kernel(
    const float* __restrict__ src, bf16_t* __restrict__ dst, int n8)
{
  const int i = blockIdx.x * 256 + threadIdx.x;
  if (i < n8) {
    const v4f a = *(const v4f*)(src + (size_t)i * 8);
    const v4f b = *(const v4f*)(src + (size_t)i * 8 + 4);
    H8 t;
#pragma unroll
    for (int j = 0; j < 4; ++j) { t.h[j] = (bf16_t)a[j]; t.h[4 + j] = (bf16_t)b[j]; }
    *(v4u*)(dst + (size_t)i * 8) = t.u;
  }
}

// ---------------------------------------------------------------------------
// Kernel 1: fused QKV projection (all-bf16 operands).
// grid = (ROWS/128, 3*NHEADS); block = 256 (8 waves). Weight tile (64 x 32)
// async-copied to double-buffered LDS (pitch 40 bf16 = 80 B: 16-B aligned
// ds_load_b128, 16 distinct banks across the fragment's 16 rows).
// Q,K written [b][h][s][d]; V written transposed [b][h][d][s] (packed b128).
// ---------------------------------------------------------------------------
#define WPITCH 40

__global__ __launch_bounds__(256) void qkv_proj_kernel(
    const bf16_t* __restrict__ X,
    const bf16_t* __restrict__ Wq, const bf16_t* __restrict__ Wk,
    const bf16_t* __restrict__ Wv,
    bf16_t* __restrict__ qo, bf16_t* __restrict__ ko, bf16_t* __restrict__ vto)
{
  __shared__ bf16_t lds_w[2][64 * WPITCH];
  const int tid  = threadIdx.x;
  const int wave = tid >> 5;
  const int lane = tid & 31;
  const int n    = lane & 15;
  const int half = lane >> 4;

  const int rowBase = blockIdx.x * 128;
  const int yy      = blockIdx.y;
  const int which   = yy / NHEADS;            // 0=Q 1=K 2=V
  const int h       = yy - which * NHEADS;
  const int colBase = h * HDIM;

  const bf16_t* W = (which == 0) ? Wq : (which == 1) ? Wk : Wv;

  const bf16_t* aPtr = X + (size_t)(rowBase + wave * 16 + n) * EMBED;

  v8f acc[4];
#pragma unroll
  for (int i = 0; i < 4; ++i) acc[i] = v8f{};

  // staging map: thread t copies one 16-B chunk per k-step
  const int     stRow = tid >> 2;
  const int     stCol = (tid & 3) * 8;
  const bf16_t* wSrc  = W + (size_t)(colBase + stRow) * EMBED + stCol;
  bf16_t*       wDst[2] = { &lds_w[0][stRow * WPITCH + stCol],
                            &lds_w[1][stRow * WPITCH + stCol] };

  async_cp16(wDst[0], wSrc);                  // prologue: tile kt=0

  for (int kt = 0; kt < NKT; ++kt) {
    const int k0 = kt * 32;
    async_wait0();
    __syncthreads();                          // tile kt resident for all waves
    if (kt + 1 < NKT)
      async_cp16(wDst[(kt + 1) & 1], wSrc + (kt + 1) * 32);

    __builtin_prefetch(aPtr + k0 + 32, 0, 1); // next A run
    const bf16_t* buf = lds_w[kt & 1];

    v16bf af = frag_ld_bf16(aPtr + k0 + 8 * half, aPtr + k0 + 16 + 8 * half);
#pragma unroll
    for (int nt = 0; nt < 4; ++nt) {
      const bf16_t* bp = &buf[(nt * 16 + n) * WPITCH];
      v16bf bfr = frag_ld_bf16(bp + 8 * half, bp + 16 + 8 * half);
      acc[nt] = wmma_bf16(af, bfr, acc[nt]);
    }
  }

  if (which == 2) {
    // V^T: a lane's 8 row-values are 8 consecutive seq positions -> one b128
    const int g0 = rowBase + wave * 16 + 8 * half;
    const int b  = g0 >> 12;
    const int s0 = g0 & (SEQ - 1);
#pragma unroll
    for (int nt = 0; nt < 4; ++nt) {
      H8 t;
#pragma unroll
      for (int r = 0; r < 8; ++r) t.h[r] = (bf16_t)acc[nt][r];
      *(v4u*)&vto[((size_t)(b * NHEADS + h) * HDIM + nt * 16 + n) * SEQ + s0] = t.u;
    }
  } else {
    bf16_t* dst = (which == 0) ? qo : ko;
#pragma unroll
    for (int nt = 0; nt < 4; ++nt) {
#pragma unroll
      for (int r = 0; r < 8; ++r) {
        const int g = rowBase + wave * 16 + r + 8 * half;
        const int b = g >> 12;
        const int s = g & (SEQ - 1);
        dst[((size_t)(b * NHEADS + h) * SEQ + s) * HDIM + nt * 16 + n] =
            (bf16_t)acc[nt][r];
      }
    }
  }
}

// ---------------------------------------------------------------------------
// Kernel 2: flash attention. grid = (SEQ/64, BATCH*NHEADS); block = 128
// (4 waves, 16 query rows each). The shared 64x64 K and V tiles are
// async-staged once per block into double-buffered LDS (pitch 72 = 144 B,
// 16-B aligned) instead of 4x-redundant global reads. Online exp2-domain
// softmax; row stats reduced with shfl_xor(width=16) to match the WMMA
// D-layout half split.
// ---------------------------------------------------------------------------
#define KVP    72
#define PPITCH 72

__global__ __launch_bounds__(128) void flash_attn_kernel(
    const bf16_t* __restrict__ q, const bf16_t* __restrict__ k,
    const bf16_t* __restrict__ vT, const float* __restrict__ mask,
    bf16_t* __restrict__ ctx)
{
  __shared__ bf16_t kbuf[2][64 * KVP];
  __shared__ bf16_t vbuf[2][64 * KVP];
  __shared__ bf16_t lds_p[4 * 16 * PPITCH];

  const int tid  = threadIdx.x;
  const int wave = tid >> 5;
  const int lane = tid & 31;
  const int n    = lane & 15;
  const int half = lane >> 4;

  const int bh = blockIdx.y;
  const int b  = bh / NHEADS;
  const int h  = bh - b * NHEADS;
  const int q0 = blockIdx.x * 64 + wave * 16;

  const bf16_t* qp = q  + (size_t)bh * SEQ * HDIM;
  const bf16_t* kp = k  + (size_t)bh * SEQ * HDIM;
  const bf16_t* vp = vT + (size_t)bh * HDIM * SEQ;
  const float*  mp = mask + (size_t)b * SEQ;

  // Q A-fragments held in registers for the whole pass (d in two 32-steps)
  const bf16_t* qr = qp + (size_t)(q0 + n) * HDIM;
  const v16bf qa0 = frag_ld_bf16(qr + 8 * half,      qr + 16 + 8 * half);
  const v16bf qa1 = frag_ld_bf16(qr + 32 + 8 * half, qr + 48 + 8 * half);

  v8f cacc[4];
#pragma unroll
  for (int i = 0; i < 4; ++i) cacc[i] = v8f{};
  float mrow[8], lrow[8];
#pragma unroll
  for (int r = 0; r < 8; ++r) { mrow[r] = -3.0e38f; lrow[r] = 0.0f; }

  bf16_t* pl = &lds_p[wave * 16 * PPITCH];
  const float scale = 0.125f;                      // 1/sqrt(HDIM)
  const float LOG2E = 1.44269504088896340736f;

  // prologue: stage K/V tile jt=0 (each thread: 4 chunks K + 4 chunks V)
#pragma unroll
  for (int i = 0; i < 4; ++i) {
    const int c   = tid + 128 * i;
    const int row = c >> 3;
    const int col = (c & 7) * 8;
    async_cp16(&kbuf[0][row * KVP + col], kp + (size_t)row * HDIM + col);
    async_cp16(&vbuf[0][row * KVP + col], vp + (size_t)row * SEQ + col);
  }

  for (int jt = 0; jt < NJT; ++jt) {
    const int jb = jt * 64;
    async_wait0();
    __syncthreads();                               // tiles jt resident
    if (jt + 1 < NJT) {
      const int nb  = (jt + 1) & 1;
      const int jbn = jb + 64;
#pragma unroll
      for (int i = 0; i < 4; ++i) {
        const int c   = tid + 128 * i;
        const int row = c >> 3;
        const int col = (c & 7) * 8;
        async_cp16(&kbuf[nb][row * KVP + col],
                   kp + (size_t)(jbn + row) * HDIM + col);
        async_cp16(&vbuf[nb][row * KVP + col],
                   vp + (size_t)row * SEQ + jbn + col);
      }
    }
    const bf16_t* kt_ = kbuf[jt & 1];
    const bf16_t* vt_ = vbuf[jt & 1];

    // ---- scores S = (Q K^T) * scale + mask : 16 x 64 in 4 accumulators ----
    v8f sacc[4];
#pragma unroll
    for (int c = 0; c < 4; ++c) {
      const bf16_t* kr = &kt_[(c * 16 + n) * KVP];
      v16bf kb0 = frag_ld_bf16(kr + 8 * half,      kr + 16 + 8 * half);
      v16bf kb1 = frag_ld_bf16(kr + 32 + 8 * half, kr + 48 + 8 * half);
      v8f s = v8f{};
      s = wmma_bf16(qa0, kb0, s);
      s = wmma_bf16(qa1, kb1, s);
      const float mv = mp[jb + c * 16 + n];
#pragma unroll
      for (int r = 0; r < 8; ++r) s[r] = s[r] * scale + mv;
      sacc[c] = s;
    }

    // ---- online softmax (exp2 domain) ----
    float tmax[8];
#pragma unroll
    for (int r = 0; r < 8; ++r)
      tmax[r] = fmaxf(fmaxf(sacc[0][r], sacc[1][r]),
                      fmaxf(sacc[2][r], sacc[3][r]));
#pragma unroll
    for (int off = 8; off >= 1; off >>= 1)
#pragma unroll
      for (int r = 0; r < 8; ++r)
        tmax[r] = fmaxf(tmax[r], __shfl_xor(tmax[r], off, 16));

    float corr[8];
#pragma unroll
    for (int r = 0; r < 8; ++r) {
      const float mnew = fmaxf(mrow[r], tmax[r]);
      corr[r] = exp2f((mrow[r] - mnew) * LOG2E);
      mrow[r] = mnew;
    }

    float rsum[8];
#pragma unroll
    for (int r = 0; r < 8; ++r) rsum[r] = 0.0f;
#pragma unroll
    for (int c = 0; c < 4; ++c)
#pragma unroll
      for (int r = 0; r < 8; ++r) {
        const float p = exp2f((sacc[c][r] - mrow[r]) * LOG2E);
        sacc[c][r] = p;
        rsum[r] += p;
      }
#pragma unroll
    for (int off = 8; off >= 1; off >>= 1)
#pragma unroll
      for (int r = 0; r < 8; ++r)
        rsum[r] += __shfl_xor(rsum[r], off, 16);
#pragma unroll
    for (int r = 0; r < 8; ++r) lrow[r] = lrow[r] * corr[r] + rsum[r];
#pragma unroll
    for (int c = 0; c < 4; ++c)
#pragma unroll
      for (int r = 0; r < 8; ++r) cacc[c][r] *= corr[r];

    // ---- P (D layout) -> per-wave LDS tile -> A-layout fragments ----
#pragma unroll
    for (int c = 0; c < 4; ++c)
#pragma unroll
      for (int r = 0; r < 8; ++r)
        pl[(r + 8 * half) * PPITCH + c * 16 + n] = (bf16_t)sacc[c][r];

    const bf16_t* pr = pl + n * PPITCH;   // same-wave RAW: dscnt wait by compiler
    v16bf pa0 = frag_ld_bf16(pr + 8 * half,      pr + 16 + 8 * half);
    v16bf pa1 = frag_ld_bf16(pr + 32 + 8 * half, pr + 48 + 8 * half);

    // ---- ctx += P * V ----
#pragma unroll
    for (int dt = 0; dt < 4; ++dt) {
      const bf16_t* vr = &vt_[(dt * 16 + n) * KVP];
      v16bf vb0 = frag_ld_bf16(vr + 8 * half,      vr + 16 + 8 * half);
      v16bf vb1 = frag_ld_bf16(vr + 32 + 8 * half, vr + 48 + 8 * half);
      v8f a = cacc[dt];
      a = wmma_bf16(pa0, vb0, a);
      a = wmma_bf16(pa1, vb1, a);
      cacc[dt] = a;
    }
  }

  // ---- finalize: ctx[b][s][h*64+d] = acc / l ----
#pragma unroll
  for (int r = 0; r < 8; ++r) {
    const float inv = 1.0f / lrow[r];
    const int s = q0 + r + 8 * half;
#pragma unroll
    for (int dt = 0; dt < 4; ++dt) {
      const int e = h * HDIM + dt * 16 + n;
      ctx[((size_t)(b * SEQ + s)) * EMBED + e] = (bf16_t)(cacc[dt][r] * inv);
    }
  }
}

// ---------------------------------------------------------------------------
// Kernel 3: output projection out = ctx @ Wo^T + bo (fp32 out).
// grid = (ROWS/128, EMBED/64); block = 256 (8 waves). Same async-LDS
// double-buffered weight staging as kernel 1.
// ---------------------------------------------------------------------------
__global__ __launch_bounds__(256) void out_proj_kernel(
    const bf16_t* __restrict__ ctx, const bf16_t* __restrict__ Wo,
    const float* __restrict__ bo, float* __restrict__ out)
{
  __shared__ bf16_t lds_w[2][64 * WPITCH];
  const int tid  = threadIdx.x;
  const int wave = tid >> 5;
  const int lane = tid & 31;
  const int n    = lane & 15;
  const int half = lane >> 4;

  const int rowBase = blockIdx.x * 128;
  const int colBase = blockIdx.y * 64;

  const bf16_t* aPtr = ctx + (size_t)(rowBase + wave * 16 + n) * EMBED;

  v8f acc[4];
#pragma unroll
  for (int i = 0; i < 4; ++i) acc[i] = v8f{};

  const int     stRow = tid >> 2;
  const int     stCol = (tid & 3) * 8;
  const bf16_t* wSrc  = Wo + (size_t)(colBase + stRow) * EMBED + stCol;
  bf16_t*       wDst[2] = { &lds_w[0][stRow * WPITCH + stCol],
                            &lds_w[1][stRow * WPITCH + stCol] };

  async_cp16(wDst[0], wSrc);

  for (int kt = 0; kt < NKT; ++kt) {
    const int k0 = kt * 32;
    async_wait0();
    __syncthreads();
    if (kt + 1 < NKT)
      async_cp16(wDst[(kt + 1) & 1], wSrc + (kt + 1) * 32);

    __builtin_prefetch(aPtr + k0 + 32, 0, 1);
    const bf16_t* buf = lds_w[kt & 1];

    v16bf af = frag_ld_bf16(aPtr + k0 + 8 * half, aPtr + k0 + 16 + 8 * half);
#pragma unroll
    for (int nt = 0; nt < 4; ++nt) {
      const bf16_t* bp = &buf[(nt * 16 + n) * WPITCH];
      v16bf bfr = frag_ld_bf16(bp + 8 * half, bp + 16 + 8 * half);
      acc[nt] = wmma_bf16(af, bfr, acc[nt]);
    }
  }

#pragma unroll
  for (int nt = 0; nt < 4; ++nt) {
    const float bias = bo[colBase + nt * 16 + n];
#pragma unroll
    for (int r = 0; r < 8; ++r) {
      const int g = rowBase + wave * 16 + r + 8 * half;
      out[(size_t)g * EMBED + colBase + nt * 16 + n] = acc[nt][r] + bias;
    }
  }
}

// ---------------------------------------------------------------------------
// Launch. Inputs: hidden_states, attention_mask, Wq, Wk, Wv, Wo, bo (fp32).
// Workspace (bf16): X, Wq, Wk, Wv, Wo, Q[b][h][s][d], K[b][h][s][d],
//                   V^T[b][h][d][s], ctx[b][s][e]  -> ~68 MB total.
// ---------------------------------------------------------------------------
extern "C" void kernel_launch(void* const* d_in, const int* in_sizes, int n_in,
                              void* d_out, int out_size, void* d_ws, size_t ws_size,
                              hipStream_t stream) {
  const float* hidden = (const float*)d_in[0];
  const float* mask   = (const float*)d_in[1];
  const float* Wq     = (const float*)d_in[2];
  const float* Wk     = (const float*)d_in[3];
  const float* Wv     = (const float*)d_in[4];
  const float* Wo     = (const float*)d_in[5];
  const float* bo     = (const float*)d_in[6];
  float* out = (float*)d_out;

  bf16_t* ws = (bf16_t*)d_ws;
  const size_t RXE = (size_t)ROWS * EMBED;       // 6,291,456
  const size_t WSZ = (size_t)EMBED * EMBED;      //   589,824
  bf16_t* xbf   = ws;
  bf16_t* wqbf  = xbf + RXE;
  bf16_t* wkbf  = wqbf + WSZ;
  bf16_t* wvbf  = wkbf + WSZ;
  bf16_t* wobf  = wvbf + WSZ;
  bf16_t* qws   = wobf + WSZ;
  bf16_t* kws   = qws + RXE;
  bf16_t* vtws  = kws + RXE;
  bf16_t* ctxws = vtws + RXE;

  // K0: one-shot bf16 conversion (8 elts/thread)
  cvt_bf16_kernel<<<(int)(RXE / 8 / 256), 256, 0, stream>>>(hidden, xbf, (int)(RXE / 8));
  cvt_bf16_kernel<<<(int)(WSZ / 8 / 256), 256, 0, stream>>>(Wq, wqbf, (int)(WSZ / 8));
  cvt_bf16_kernel<<<(int)(WSZ / 8 / 256), 256, 0, stream>>>(Wk, wkbf, (int)(WSZ / 8));
  cvt_bf16_kernel<<<(int)(WSZ / 8 / 256), 256, 0, stream>>>(Wv, wvbf, (int)(WSZ / 8));
  cvt_bf16_kernel<<<(int)(WSZ / 8 / 256), 256, 0, stream>>>(Wo, wobf, (int)(WSZ / 8));

  qkv_proj_kernel<<<dim3(ROWS / 128, 3 * NHEADS), 256, 0, stream>>>(
      xbf, wqbf, wkbf, wvbf, qws, kws, vtws);
  flash_attn_kernel<<<dim3(SEQ / 64, BATCH * NHEADS), 128, 0, stream>>>(
      qws, kws, vtws, mask, ctxws);
  out_proj_kernel<<<dim3(ROWS / 128, EMBED / 64), 256, 0, stream>>>(
      ctxws, wobf, bo, out);
}
